// NeRFRenderer_15333033246724
// MI455X (gfx1250) — compile-verified, matrix operand order checked by reading, not run on hardware
//
#include <hip/hip_runtime.h>
#include <hip/hip_bf16.h>
#include <math.h>

typedef __attribute__((ext_vector_type(16))) _Float16 v16h;
typedef __attribute__((ext_vector_type(8)))  float    v8f;

#define UC   128          // coarse sample count
#define HID  32           // MLP hidden width
#define RPB  4            // rays per block (1 wave32 per ray)

// ---- raw hardware transcendentals (inputs are well-scaled, no fixups) -----
__device__ __forceinline__ float fast_exp(float x) {
#if __has_builtin(__builtin_amdgcn_exp2f)
  return __builtin_amdgcn_exp2f(x * 1.44269504088896340736f);   // v_exp_f32
#else
  return __expf(x);
#endif
}
__device__ __forceinline__ float fast_log(float x) {
#if __has_builtin(__builtin_amdgcn_logf)
  return 0.69314718055994530942f * __builtin_amdgcn_logf(x);    // v_log_f32
#else
  return __logf(x);
#endif
}
__device__ __forceinline__ float fast_tanh(float x) {
#if __has_builtin(__builtin_amdgcn_tanhf)
  return __builtin_amdgcn_tanhf(x);                             // v_tanh_f32
#elif __has_builtin(__builtin_amdgcn_tanh_f32)
  return __builtin_amdgcn_tanh_f32(x);
#else
  float xc = fminf(fmaxf(x, -15.f), 15.f);
  float e  = fast_exp(2.f * xc);
  return (e - 1.f) / (e + 1.f);
#endif
}
// softplus(x) = max(x,0) + log1p(exp(-|x|)), exact and branch-free
__device__ __forceinline__ float softplus_f(float x) {
  return fmaxf(x, 0.f) + fast_log(1.f + fast_exp(-fabsf(x)));
}

// inclusive wave32 scan (Hillis-Steele)
__device__ __forceinline__ float wave_incl_scan(float x, int lane) {
  #pragma unroll
  for (int off = 1; off < 32; off <<= 1) {
    float y = __shfl_up(x, off, 32);
    if (lane >= off) x += y;
  }
  return x;
}

// Evaluate density MLP for 16 points (one tile) with WMMA.
// Layer1: pts[16x3] @ W1[3x32]  -> two v_wmma_f32_16x16x32_f16 (K padded to 32)
// Layer2: tanh(h)[16x32] @ W2[32x1] -> batched 16-lane butterfly reduction
// STORE=true : write sigma[m] to LDS.  STORE=false : acc += sigma*delta.
template<bool STORE>
__device__ __forceinline__ void density_tile(
    int base, int total, const float* __restrict__ zbuf, float sd,
    float ox, float oy, float oz, float dx, float dy, float dz,
    v16h b0v, v16h b1v, float w2a, float w2b,
    float biasA, float biasB, float b2s, int lane,
    float* sig_out, float* acc)
{
  // A-matrix (16x32 f16): lane<16 holds row M=lane, K=0..2 = (x,y,z); rest 0.
  const bool lo16 = (lane & 16) == 0;
  float zv = zbuf[base + (lane & 15)];
  v16h a;
  #pragma unroll
  for (int j = 0; j < 16; ++j) a[j] = (_Float16)0.f;
  a[0] = (_Float16)(lo16 ? (ox + dx * zv) : 0.f);
  a[1] = (_Float16)(lo16 ? (oy + dy * zv) : 0.f);
  a[2] = (_Float16)(lo16 ? (oz + dz * zv) : 0.f);

  v8f c0 = {0.f,0.f,0.f,0.f,0.f,0.f,0.f,0.f};
  v8f c1 = {0.f,0.f,0.f,0.f,0.f,0.f,0.f,0.f};
  c0 = __builtin_amdgcn_wmma_f32_16x16x32_f16(false, a, false, b0v, (short)0, c0, false, false);
  c1 = __builtin_amdgcn_wmma_f32_16x16x32_f16(false, a, false, b1v, (short)0, c1, false, false);

  // per-row layer-2 partial: v[r] = sum over this lane's 2 hidden units
  float v[8];
  #pragma unroll
  for (int r = 0; r < 8; ++r)
    v[r] = fast_tanh(c0[r] + biasA) * w2a + fast_tanh(c1[r] + biasB) * w2b;

  // butterfly over the 16-lane half-groups; stage-major so all 8 shuffles of a
  // stage issue before their waits (DS returns in order -> latency hidden)
  #pragma unroll
  for (int off = 1; off < 16; off <<= 1) {
    float t[8];
    #pragma unroll
    for (int r = 0; r < 8; ++r) t[r] = __shfl_xor(v[r], off, 32);
    #pragma unroll
    for (int r = 0; r < 8; ++r) v[r] += t[r];
  }

  // every lane now holds all 8 row-sums of its half; select own row (cndmask
  // tree), one softplus, one unconditional store (lanes l and l+8 duplicate).
  const int r3 = lane & 7;
  float s0 = (r3 & 1) ? v[1] : v[0];
  float s1 = (r3 & 1) ? v[3] : v[2];
  float s2 = (r3 & 1) ? v[5] : v[4];
  float s3 = (r3 & 1) ? v[7] : v[6];
  float t0 = (r3 & 2) ? s1 : s0;
  float t1 = (r3 & 2) ? s3 : s2;
  float sel = (r3 & 4) ? t1 : t0;
  float sig = softplus_f(sel + b2s);
  int   m   = base + r3 + ((lane >> 4) << 3);

  if (STORE) {
    sig_out[m] = sig;                       // duplicate write, same value
  } else {
    int   mp  = (m + 1 < total) ? m + 1 : m;
    float zm  = zbuf[m];
    float dlt = (m < total - 1) ? (zbuf[mp] - zm) : sd;
    float g   = (lane & 8) ? 0.f : 1.f;     // count each row once
    *acc += g * sig * dlt;
  }
}

__global__ __launch_bounds__(32 * RPB)
void nerf_transmittance_kernel(
    const float* __restrict__ rays_o, const float* __restrict__ rays_d,
    const float* __restrict__ nearv,  const float* __restrict__ farv,
    const float* __restrict__ noise,  const float* __restrict__ W1,
    const float* __restrict__ b1v_g,  const float* __restrict__ W2,
    const float* __restrict__ b2v_g,  float* __restrict__ out, int nRays)
{
  __shared__ float zc  [RPB][UC];       // coarse z (sorted by construction)
  __shared__ float sg  [RPB][UC];       // coarse sigma
  __shared__ float cdf [RPB][UC];       // cdf[0..126]
  __shared__ float nz  [RPB][UC];       // importance samples (sorted)
  __shared__ float zall[RPB][2 * UC];   // merged fine samples

  const int wid  = threadIdx.x >> 5;
  const int lane = threadIdx.x & 31;
  int ray = blockIdx.x * RPB + wid;
  if (ray >= nRays) ray = nRays - 1;    // clamp (duplicate work, deterministic)

  const float ox = rays_o[ray*3+0], oy = rays_o[ray*3+1], oz = rays_o[ray*3+2];
  const float dx = rays_d[ray*3+0], dy = rays_d[ray*3+1], dz = rays_d[ray*3+2];
  const float nr = nearv[ray], fr = farv[ray];
  const float sd = (fr - nr) * (1.0f / UC);

  // --- constant WMMA B operands: W1 columns (B 32x16: lanes0-15 hold K=0..2 = W1 rows; rest 0)
  const int  col  = lane & 15;
  const bool lo16 = (lane & 16) == 0;
  v16h b0v, b1v;
  #pragma unroll
  for (int j = 0; j < 16; ++j) { b0v[j] = (_Float16)0.f; b1v[j] = (_Float16)0.f; }
  #pragma unroll
  for (int k = 0; k < 3; ++k) {
    float wa = W1[k * HID + col];
    float wb = W1[k * HID + col + 16];
    b0v[k] = (_Float16)(lo16 ? wa : 0.f);
    b1v[k] = (_Float16)(lo16 ? wb : 0.f);
  }
  const float biasA = b1v_g[col], biasB = b1v_g[col + 16];
  const float w2a = W2[col], w2b = W2[col + 16];
  const float b2s = b2v_g[0];

  // --- 1. stratified coarse z (coalesced noise read) ---
  #pragma unroll
  for (int k = 0; k < 4; ++k) {
    int i = lane + 32 * k;
    float t = (float)i * (1.0f / (UC - 1));
    zc[wid][i] = nr + (fr - nr) * t + (noise[ray * UC + i] - 0.5f) * sd;
  }
  __syncthreads();

  // --- 2. coarse density (WMMA) ---
  #pragma unroll 2
  for (int t = 0; t < UC / 16; ++t)
    density_tile<true>(t * 16, UC, zc[wid], sd, ox,oy,oz, dx,dy,dz,
                       b0v, b1v, w2a, w2b, biasA, biasB, b2s, lane,
                       sg[wid], nullptr);
  __syncthreads();

  // --- 3. weights via log-space cumprod scan (4 contiguous elems / lane) ---
  float lg[4], al[4];
  #pragma unroll
  for (int q = 0; q < 4; ++q) {
    int i = lane * 4 + q;
    float d = (i < UC - 1) ? (zc[wid][i + 1] - zc[wid][i]) : sd;
    float e = fast_exp(-d * sg[wid][i]);        // 1 - alpha
    al[q] = 1.f - e;
    lg[q] = fast_log(e + 1e-15f);
  }
  float run = 0.f, exc[4];
  #pragma unroll
  for (int q = 0; q < 4; ++q) { exc[q] = run; run += lg[q]; }
  float incl = wave_incl_scan(run, lane);
  float off  = incl - run;                      // exclusive wave offset
  float w[4];
  #pragma unroll
  for (int q = 0; q < 4; ++q) w[q] = al[q] * fast_exp(off + exc[q]);

  // windowed cumsum over weights[1..126] (+1e-5) -> CDF
  float wp[4], runw = 0.f, exw[4];
  #pragma unroll
  for (int q = 0; q < 4; ++q) {
    int i = lane * 4 + q;
    wp[q] = (i >= 1 && i <= UC - 2) ? (w[q] + 1e-5f) : 0.f;
    exw[q] = runw; runw += wp[q];
  }
  float inclw = wave_incl_scan(runw, lane);
  float offw  = inclw - runw;
  float invS  = 1.f / __shfl(inclw, 31, 32);    // total weight
  #pragma unroll
  for (int q = 0; q < 4; ++q) {
    int i = lane * 4 + q;
    if (i <= UC - 2) cdf[wid][i] = (i == 0) ? 0.f : (offw + exw[q] + wp[q]) * invS;
  }
  __syncthreads();

  // --- 4. deterministic inverse-CDF sampling (searchsorted + lerp in z_mid) ---
  const float u0 = 0.5f / UC;
  const float du = (1.f - 1.f / UC) / (UC - 1);
  #pragma unroll
  for (int k = 0; k < 4; ++k) {
    int s = lane + 32 * k;
    float u = u0 + du * (float)s;
    int lo = 0, hi = UC - 1;                    // cdf has 127 entries
    while (lo < hi) { int mid = (lo + hi) >> 1; if (cdf[wid][mid] <= u) lo = mid + 1; else hi = mid; }
    int below = lo - 1; below = below < 0 ? 0 : below;
    int above = lo < UC - 2 ? lo : UC - 2;
    float cb = cdf[wid][below], ca = cdf[wid][above];
    float bb = 0.5f * (zc[wid][below] + zc[wid][below + 1]);   // z_mid on the fly
    float ba = 0.5f * (zc[wid][above] + zc[wid][above + 1]);
    float den = ca - cb; den = (den < 1e-5f) ? 1.f : den;
    nz[wid][s] = bb + ((u - cb) / den) * (ba - bb);
  }
  __syncthreads();

  // --- 5. merge-by-rank (both lists sorted -> no full sort needed) ---
  #pragma unroll
  for (int k = 0; k < 4; ++k) {
    int i = lane + 32 * k;
    float v = zc[wid][i];
    int lo = 0, hi = UC;
    while (lo < hi) { int mid = (lo + hi) >> 1; if (nz[wid][mid] <  v) lo = mid + 1; else hi = mid; }
    zall[wid][i + lo] = v;
    float v2 = nz[wid][i];
    lo = 0; hi = UC;
    while (lo < hi) { int mid = (lo + hi) >> 1; if (zc[wid][mid] <= v2) lo = mid + 1; else hi = mid; }
    zall[wid][i + lo] = v2;
  }
  __syncthreads();

  // --- 6. fine density (WMMA) + transmittance reduction ---
  float acc = 0.f;
  #pragma unroll 2
  for (int t = 0; t < (2 * UC) / 16; ++t)
    density_tile<false>(t * 16, 2 * UC, zall[wid], sd, ox,oy,oz, dx,dy,dz,
                        b0v, b1v, w2a, w2b, biasA, biasB, b2s, lane,
                        nullptr, &acc);
  acc += __shfl_xor(acc, 1, 32);
  acc += __shfl_xor(acc, 2, 32);
  acc += __shfl_xor(acc, 4, 32);
  acc += __shfl_xor(acc, 8, 32);
  acc += __shfl_xor(acc, 16, 32);
  if (lane == 0) out[ray] = fast_exp(-acc);
}

extern "C" void kernel_launch(void* const* d_in, const int* in_sizes, int n_in,
                              void* d_out, int out_size, void* d_ws, size_t ws_size,
                              hipStream_t stream) {
  const float* rays_o = (const float*)d_in[0];
  const float* rays_d = (const float*)d_in[1];
  const float* nearv  = (const float*)d_in[2];
  const float* farv   = (const float*)d_in[3];
  const float* noise  = (const float*)d_in[4];
  const float* W1     = (const float*)d_in[5];
  const float* b1     = (const float*)d_in[6];
  const float* W2     = (const float*)d_in[7];
  const float* b2     = (const float*)d_in[8];
  float* out = (float*)d_out;

  const int nRays  = in_sizes[2];                 // near is [N,1]
  const int blocks = (nRays + RPB - 1) / RPB;
  nerf_transmittance_kernel<<<blocks, 32 * RPB, 0, stream>>>(
      rays_o, rays_d, nearv, farv, noise, W1, b1, W2, b2, out, nRays);
}
